// Mamba_16157666967920
// MI455X (gfx1250) — matrix operand order repaired and checked by reading, not compiled, not run on hardware
//
#include <hip/hip_runtime.h>
#include <hip/hip_bf16.h>

// ---------------- problem constants ----------------
constexpr int Bsz     = 4;
constexpr int L       = 4096;
constexpr int DM      = 1024;
constexpr int DSTATE  = 16;
constexpr int NH      = 8;
constexpr int CHUNK   = 64;
constexpr int DIN     = 2048;
constexpr int HDIM    = 256;               // DIN / NH
constexpr int CONVDIM = DIN + 2 * DSTATE;  // 2080
constexpr int DPROJ   = 2 * DIN + 2 * DSTATE + NH; // 4136
constexpr int PROJ_STRIDE = 4144;          // padded to mult of 16
constexpr int NPAD1   = 4160;              // W_in rows padded to mult of 64
constexpr int NCH     = L / CHUNK;         // 64
constexpr int BL      = Bsz * L;           // 16384
constexpr int KSTEP   = 64;                // GEMM K tile (2 wmma k-steps)

typedef __attribute__((ext_vector_type(16))) __bf16 v16bf;
typedef __attribute__((ext_vector_type(8)))  float  v8f;
typedef __attribute__((ext_vector_type(4)))  int    v4i;

union BF16x16 {
    v16bf v;
    uint4 q[2];
    unsigned short u[16];
};

__device__ __forceinline__ unsigned short f2bf(float f) {
    unsigned int u = __float_as_uint(f);
    u += 0x7FFFu + ((u >> 16) & 1u);   // round-to-nearest-even
    return (unsigned short)(u >> 16);
}

// CDNA5 async global->LDS path (ASYNCcnt), if this toolchain exposes it.
#if defined(__has_builtin)
#if __has_builtin(__builtin_amdgcn_global_load_async_to_lds_b128) && \
    __has_builtin(__builtin_amdgcn_s_wait_asynccnt)
#define HAVE_ASYNC_LDS 1
#endif
#endif
#ifndef HAVE_ASYNC_LDS
#define HAVE_ASYNC_LDS 0
#endif

__device__ __forceinline__ void copy16B_to_lds(const unsigned short* src, unsigned short* dst) {
#if HAVE_ASYNC_LDS
    v4i* s = (v4i*)const_cast<unsigned short*>(src);
    v4i* d = (v4i*)dst;
    __builtin_amdgcn_global_load_async_to_lds_b128(
        (__attribute__((address_space(1))) v4i*)s,
        (__attribute__((address_space(3))) v4i*)d, 0, 0);
#else
    *(uint4*)dst = *(const uint4*)src;
#endif
}

// ---------------- f32 -> bf16 convert (with zero-pad tail) ----------------
__global__ __launch_bounds__(256)
void cvt_bf16_kernel(const float* __restrict__ src, unsigned short* __restrict__ dst,
                     size_t srcCount, size_t dstCount) {
    size_t t = (size_t)blockIdx.x * blockDim.x + threadIdx.x;
    if (t >= dstCount) return;
    dst[t] = (t < srcCount) ? f2bf(src[t]) : (unsigned short)0;
}

// ---------------- bf16 WMMA GEMM:  C[M,Nreal] = A[M,K] * Bw[N,K]^T ----------------
// Block tile 128(M) x 64(N), 8 waves, wave = 16 rows x 64 cols (4 wmma tiles).
// Double-buffered LDS with async global->LDS copies (ASYNCcnt pipeline).
__global__ __launch_bounds__(256)
void gemm_bf16_kernel(const unsigned short* __restrict__ A,
                      const unsigned short* __restrict__ Bw,
                      float* __restrict__ C,
                      int K, int ldc, int Nreal) {
    __shared__ __attribute__((aligned(16))) unsigned short As[2][128 * KSTEP]; // 2x16KB
    __shared__ __attribute__((aligned(16))) unsigned short Bs[2][64 * KSTEP];  // 2x8KB

    const int tid  = threadIdx.x;
    const int wave = tid >> 5;
    const int lane = tid & 31;
    const int lh   = lane >> 4;   // 0/1: K-half select
    const int ll   = lane & 15;
    const int m0   = blockIdx.y * 128;
    const int n0   = blockIdx.x * 64;

    v8f acc[4] = {};

    // Per-thread copy slots (6 x 16B async ops / thread / stage):
    //   A tile 128x64: row = tid>>1, 32 elems at (tid&1)*32   -> 4 x b128
    //   B tile  64x64: row = tid>>2, 16 elems at (tid&3)*16   -> 2 x b128
    const int ra = tid >> 1, ha = (tid & 1) * 32;
    const int rb = tid >> 2, hb = (tid & 3) * 16;

    auto issue = [&](int buf, int k0) {
        const unsigned short* sa = A + (size_t)(m0 + ra) * K + k0 + ha;
        unsigned short* da = &As[buf][ra * KSTEP + ha];
#pragma unroll
        for (int j = 0; j < 4; j++) copy16B_to_lds(sa + j * 8, da + j * 8);
        const unsigned short* sb = Bw + (size_t)(n0 + rb) * K + k0 + hb;
        unsigned short* db = &Bs[buf][rb * KSTEP + hb];
#pragma unroll
        for (int j = 0; j < 2; j++) copy16B_to_lds(sb + j * 8, db + j * 8);
    };

    const int nsteps = K / KSTEP;
    issue(0, 0);
    for (int s = 0; s < nsteps; s++) {
        const int cur = s & 1;
        if (s + 1 < nsteps) {
            issue(cur ^ 1, (s + 1) * KSTEP);
#if HAVE_ASYNC_LDS
            __builtin_amdgcn_s_wait_asynccnt(6);  // only next stage's 6 ops may remain
#endif
        } else {
#if HAVE_ASYNC_LDS
            __builtin_amdgcn_s_wait_asynccnt(0);
#endif
        }
        __syncthreads();

        const unsigned short* Ab = &As[cur][0];
        const unsigned short* Bb = &Bs[cur][0];
#pragma unroll
        for (int ks = 0; ks < 2; ks++) {
            BF16x16 af;   // A frag: row = wave*16+ll, K half = lh*16
            {
                int row = wave * 16 + ll;
                int kb  = ks * 32 + lh * 16;
                af.q[0] = *(const uint4*)(&Ab[row * KSTEP + kb]);
                af.q[1] = *(const uint4*)(&Ab[row * KSTEP + kb + 8]);
            }
            BF16x16 bf[4];  // B frags: col = nt*16+ll, K half = lh*16
#pragma unroll
            for (int nt = 0; nt < 4; nt++) {
                int col = nt * 16 + ll;
                int kb2 = ks * 32 + lh * 16;
                bf[nt].q[0] = *(const uint4*)(&Bb[col * KSTEP + kb2]);
                bf[nt].q[1] = *(const uint4*)(&Bb[col * KSTEP + kb2 + 8]);
            }
#pragma unroll
            for (int nt = 0; nt < 4; nt++)
                acc[nt] = __builtin_amdgcn_wmma_f32_16x16x32_bf16(
                    false, af.v, false, bf[nt].v, (short)0, acc[nt], false, false);
        }
        __syncthreads();
    }

#pragma unroll
    for (int nt = 0; nt < 4; nt++) {
        int col = n0 + nt * 16 + ll;
        if (col < Nreal) {
#pragma unroll
            for (int v = 0; v < 8; v++) {
                int row = m0 + wave * 16 + v + lh * 8;
                C[(size_t)row * ldc + col] = acc[nt][v];
            }
        }
    }
}

// ---------------- dt = softplus(proj_dt + dt_bias) ----------------
__global__ __launch_bounds__(256)
void dt_kernel(const float* __restrict__ proj, const float* __restrict__ dtb,
               float* __restrict__ dt) {
    int t = blockIdx.x * blockDim.x + threadIdx.x;
    if (t >= BL * NH) return;
    int h = t & 7;
    int bl = t >> 3;
    float v = proj[(size_t)bl * PROJ_STRIDE + DIN + CONVDIM + h] + dtb[h];
    dt[t] = (v > 20.f) ? v : log1pf(__expf(v));
}

// ---------------- depthwise causal conv (k=4) + SiLU ----------------
__global__ __launch_bounds__(256)
void dwconv_silu_kernel(const float* __restrict__ proj, const float* __restrict__ cw,
                        const float* __restrict__ cb, float* __restrict__ xbc) {
    size_t t = (size_t)blockIdx.x * blockDim.x + threadIdx.x;
    if (t >= (size_t)BL * CONVDIM) return;
    int ch = (int)(t % CONVDIM);
    size_t bl = t / CONVDIM;
    int l = (int)(bl % L);
    float acc = cb[ch];
#pragma unroll
    for (int k = 0; k < 4; k++) {
        int li = l - 3 + k;
        if (li >= 0)
            acc += cw[ch * 4 + k] * proj[(bl + (size_t)(li - l)) * PROJ_STRIDE + DIN + ch];
    }
    xbc[bl * CONVDIM + ch] = acc / (1.f + __expf(-acc));
}

// ---------------- SSD per-chunk: Y_diag (+D skip) and chunk states ----------------
__global__ __launch_bounds__(256)
void ssd_chunk_kernel(const float* __restrict__ xbc, const float* __restrict__ dt,
                      const float* __restrict__ logA, const float* __restrict__ Dskip,
                      float* __restrict__ ybuf, float* __restrict__ states) {
    const int c = blockIdx.x, b = blockIdx.y, h = blockIdx.z;
    const int tid = threadIdx.x;
    const float Ah = -__expf(logA[h]);
    const size_t blBase = (size_t)(b * L + c * CHUNK);

    __shared__ float Cs[CHUNK][DSTATE];
    __shared__ float Bsm[CHUNK][DSTATE];
    __shared__ float dts[CHUNK];
    __shared__ __attribute__((aligned(16))) unsigned short Mbf[CHUNK][CHUNK];   // 8KB
    __shared__ __attribute__((aligned(16))) unsigned short xhT[HDIM][CHUNK];    // 32KB
    __shared__ __attribute__((aligned(16))) unsigned short Bwbf[DSTATE][CHUNK]; // 2KB

    for (int i = tid; i < CHUNK * DSTATE; i += 256) {
        int l = i >> 4, n = i & 15;
        const float* row = xbc + (blBase + l) * CONVDIM + DIN;
        Bsm[l][n] = row[n];
        Cs[l][n]  = row[DSTATE + n];
    }
    if (tid < CHUNK) dts[tid] = dt[(blBase + tid) * NH + h];
    __syncthreads();

    // masked decay-weighted Gram matrix  M[l][s] = (C_l . B_s) * exp(A*(l-s)), s<=l
    for (int i = tid; i < CHUNK * CHUNK; i += 256) {
        int l = i >> 6, s = i & 63;
        float g = 0.f;
        if (s <= l) {
#pragma unroll
            for (int n = 0; n < DSTATE; n++) g += Cs[l][n] * Bsm[s][n];
            g *= __expf(Ah * (float)(l - s));
        }
        Mbf[l][s] = f2bf(g);
    }
    // Bweighted^T[n][l] = B[l][n]*exp(A*(63-l))
    for (int i = tid; i < DSTATE * CHUNK; i += 256) {
        int n = i >> 6, l = i & 63;
        Bwbf[n][l] = f2bf(Bsm[l][n] * __expf(Ah * (float)(CHUNK - 1 - l)));
    }
    // xh^T[p][l] = x[l][p]*dt[l]
    for (int i = tid; i < CHUNK * HDIM; i += 256) {
        int l = i >> 8, p = i & 255;
        xhT[p][l] = f2bf(xbc[(blBase + l) * CONVDIM + h * HDIM + p] * dts[l]);
    }
    __syncthreads();

    const int wave = tid >> 5, lane = tid & 31;
    const int lh = lane >> 4, ll = lane & 15;
    const float Dh = Dskip[h];

    // Y_diag = M(64x64) @ xh(64x256): 4x16 tiles, 8 per wave
    for (int t8 = 0; t8 < 8; t8++) {
        int tileid = t8 * 8 + wave;
        int mt = tileid & 3, nt = tileid >> 2;
        v8f acc = {};
#pragma unroll
        for (int ks = 0; ks < 2; ks++) {
            BF16x16 af, bf;
            int arow = mt * 16 + ll, kb = ks * 32 + lh * 16;
            af.q[0] = *(const uint4*)(&Mbf[arow][kb]);
            af.q[1] = *(const uint4*)(&Mbf[arow][kb + 8]);
            int bcol = nt * 16 + ll, kb2 = ks * 32 + lh * 16;
            bf.q[0] = *(const uint4*)(&xhT[bcol][kb2]);
            bf.q[1] = *(const uint4*)(&xhT[bcol][kb2 + 8]);
            acc = __builtin_amdgcn_wmma_f32_16x16x32_bf16(
                false, af.v, false, bf.v, (short)0, acc, false, false);
        }
#pragma unroll
        for (int v = 0; v < 8; v++) {
            int l = mt * 16 + v + lh * 8;
            int p = nt * 16 + ll;
            float xv = xbc[(blBase + l) * CONVDIM + h * HDIM + p] * dts[l];
            ybuf[(blBase + l) * DIN + h * HDIM + p] = acc[v] + xv * Dh;
        }
    }

    // states = Bw(16x64) @ xh(64x256): 16 p-tiles, 2 per wave
    size_t sbase = (((size_t)(b * NH + h) * NCH + c) * DSTATE) * HDIM;
    for (int t2 = 0; t2 < 2; t2++) {
        int nt = wave * 2 + t2;
        v8f acc = {};
#pragma unroll
        for (int ks = 0; ks < 2; ks++) {
            BF16x16 af, bf;
            int kb = ks * 32 + lh * 16;
            af.q[0] = *(const uint4*)(&Bwbf[ll][kb]);
            af.q[1] = *(const uint4*)(&Bwbf[ll][kb + 8]);
            int bcol = nt * 16 + ll, kb2 = ks * 32 + lh * 16;
            bf.q[0] = *(const uint4*)(&xhT[bcol][kb2]);
            bf.q[1] = *(const uint4*)(&xhT[bcol][kb2 + 8]);
            acc = __builtin_amdgcn_wmma_f32_16x16x32_bf16(
                false, af.v, false, bf.v, (short)0, acc, false, false);
        }
#pragma unroll
        for (int v = 0; v < 8; v++) {
            int n = v + lh * 8;
            int p = nt * 16 + ll;
            states[sbase + (size_t)n * HDIM + p] = acc[v];
        }
    }
}

// ---------------- sequential inter-chunk recurrence per (b,h) ----------------
__global__ __launch_bounds__(256)
void ssd_scan_kernel(const float* __restrict__ states, float* __restrict__ prev,
                     const float* __restrict__ logA) {
    const int bh = blockIdx.x;
    const int h = bh & 7;
    const float decay = __expf(-__expf(logA[h]) * (float)CHUNK);
    const int tid = threadIdx.x;
    float S[16];
#pragma unroll
    for (int i = 0; i < 16; i++) S[i] = 0.f;
    size_t base = (size_t)bh * NCH * DSTATE * HDIM;
    for (int c = 0; c < NCH; c++) {
        size_t off = base + (size_t)c * DSTATE * HDIM + (size_t)tid * 16;
#pragma unroll
        for (int i = 0; i < 16; i++) {
            prev[off + i] = S[i];
            S[i] = S[i] * decay + states[off + i];
        }
    }
}

// ---------------- Y_off = exp(Acs)*C @ state_prev, accumulated into ybuf ----------------
__global__ __launch_bounds__(256)
void ssd_yoff_kernel(const float* __restrict__ xbc, const float* __restrict__ prev,
                     const float* __restrict__ logA, float* __restrict__ ybuf) {
    const int c = blockIdx.x, b = blockIdx.y, h = blockIdx.z;
    const int tid = threadIdx.x;
    const float Ah = -__expf(logA[h]);
    const size_t blBase = (size_t)(b * L + c * CHUNK);

    __shared__ __attribute__((aligned(16))) unsigned short Cebf[CHUNK][32]; // K padded 16->32
    __shared__ __attribute__((aligned(16))) unsigned short pT[HDIM][32];

    for (int i = tid; i < CHUNK * 32; i += 256) {
        int l = i >> 5, n = i & 31;
        float v = 0.f;
        if (n < DSTATE)
            v = xbc[(blBase + l) * CONVDIM + DIN + DSTATE + n] * __expf(Ah * (float)(l + 1));
        Cebf[l][n] = f2bf(v);
    }
    size_t pbase = (((size_t)(b * NH + h) * NCH + c) * DSTATE) * HDIM;
    for (int i = tid; i < HDIM * 32; i += 256) {
        int p = i >> 5, n = i & 31;
        float v = (n < DSTATE) ? prev[pbase + (size_t)n * HDIM + p] : 0.f;
        pT[p][n] = f2bf(v);
    }
    __syncthreads();

    const int wave = tid >> 5, lane = tid & 31;
    const int lh = lane >> 4, ll = lane & 15;
    for (int t8 = 0; t8 < 8; t8++) {
        int tileid = t8 * 8 + wave;
        int mt = tileid & 3, nt = tileid >> 2;
        BF16x16 af, bf;
        int arow = mt * 16 + ll;
        af.q[0] = *(const uint4*)(&Cebf[arow][lh * 16]);
        af.q[1] = *(const uint4*)(&Cebf[arow][lh * 16 + 8]);
        int bcol = nt * 16 + ll;
        bf.q[0] = *(const uint4*)(&pT[bcol][lh * 16]);
        bf.q[1] = *(const uint4*)(&pT[bcol][lh * 16 + 8]);
        v8f acc = {};
        acc = __builtin_amdgcn_wmma_f32_16x16x32_bf16(
            false, af.v, false, bf.v, (short)0, acc, false, false);
#pragma unroll
        for (int v = 0; v < 8; v++) {
            int l = mt * 16 + v + lh * 8;
            int p = nt * 16 + ll;
            ybuf[(blBase + l) * DIN + h * HDIM + p] += acc[v];
        }
    }
}

// ---------------- gate with SiLU(z), RMSNorm, emit bf16 for GEMM2 ----------------
__global__ __launch_bounds__(256)
void gate_norm_kernel(const float* __restrict__ ybuf, const float* __restrict__ proj,
                      const float* __restrict__ normw, unsigned short* __restrict__ ybf) {
    const int bl = blockIdx.x;
    const int tid = threadIdx.x;
    __shared__ float red[256];
    float vals[8];
    float ss = 0.f;
#pragma unroll
    for (int j = 0; j < 8; j++) {
        int d = tid * 8 + j;
        float y = ybuf[(size_t)bl * DIN + d];
        float z = proj[(size_t)bl * PROJ_STRIDE + d];
        y *= z / (1.f + __expf(-z));
        vals[j] = y;
        ss += y * y;
    }
    red[tid] = ss;
    __syncthreads();
    for (int s = 128; s > 0; s >>= 1) {
        if (tid < s) red[tid] += red[tid + s];
        __syncthreads();
    }
    float scale = rsqrtf(red[0] / (float)DIN + 1e-5f);
#pragma unroll
    for (int j = 0; j < 8; j++) {
        int d = tid * 8 + j;
        ybf[(size_t)bl * DIN + d] = f2bf(vals[j] * scale * normw[d]);
    }
}

// ---------------- host ----------------
extern "C" void kernel_launch(void* const* d_in, const int* in_sizes, int n_in,
                              void* d_out, int out_size, void* d_ws, size_t ws_size,
                              hipStream_t stream) {
    const float* inp   = (const float*)d_in[0];
    const float* Win   = (const float*)d_in[1];
    const float* convw = (const float*)d_in[2];
    const float* convb = (const float*)d_in[3];
    const float* dtb   = (const float*)d_in[4];
    const float* logA  = (const float*)d_in[5];
    const float* Dskip = (const float*)d_in[6];
    const float* normw = (const float*)d_in[7];
    const float* Wout  = (const float*)d_in[8];
    float* out = (float*)d_out;

    size_t off = 0;
    auto alloc = [&](size_t bytes) {
        void* p = (char*)d_ws + off;
        off += (bytes + 255) & ~(size_t)255;
        return p;
    };
    unsigned short* inA_bf  = (unsigned short*)alloc((size_t)BL * DM * 2);
    unsigned short* win_bf  = (unsigned short*)alloc((size_t)NPAD1 * DM * 2);
    unsigned short* wout_bf = (unsigned short*)alloc((size_t)DM * DIN * 2);
    float* proj   = (float*)alloc((size_t)BL * PROJ_STRIDE * 4);
    float* xbc    = (float*)alloc((size_t)BL * CONVDIM * 4);
    float* dt     = (float*)alloc((size_t)BL * NH * 4);
    float* states = (float*)alloc((size_t)Bsz * NH * NCH * DSTATE * HDIM * 4);
    float* prev   = (float*)alloc((size_t)Bsz * NH * NCH * DSTATE * HDIM * 4);
    float* ybuf   = (float*)alloc((size_t)BL * DIN * 4);
    unsigned short* ybf = (unsigned short*)alloc((size_t)BL * DIN * 2);

    auto blocks = [](size_t n, int bs) { return (unsigned)((n + bs - 1) / bs); };

    // converts
    cvt_bf16_kernel<<<blocks((size_t)BL * DM, 256), 256, 0, stream>>>(
        inp, inA_bf, (size_t)BL * DM, (size_t)BL * DM);
    cvt_bf16_kernel<<<blocks((size_t)NPAD1 * DM, 256), 256, 0, stream>>>(
        Win, win_bf, (size_t)DPROJ * DM, (size_t)NPAD1 * DM);
    cvt_bf16_kernel<<<blocks((size_t)DM * DIN, 256), 256, 0, stream>>>(
        Wout, wout_bf, (size_t)DM * DIN, (size_t)DM * DIN);

    // GEMM1: proj = input @ W_in^T
    gemm_bf16_kernel<<<dim3(NPAD1 / 64, BL / 128), 256, 0, stream>>>(
        inA_bf, win_bf, proj, DM, PROJ_STRIDE, DPROJ);

    dt_kernel<<<blocks((size_t)BL * NH, 256), 256, 0, stream>>>(proj, dtb, dt);
    dwconv_silu_kernel<<<blocks((size_t)BL * CONVDIM, 256), 256, 0, stream>>>(
        proj, convw, convb, xbc);

    ssd_chunk_kernel<<<dim3(NCH, Bsz, NH), 256, 0, stream>>>(
        xbc, dt, logA, Dskip, ybuf, states);
    ssd_scan_kernel<<<Bsz * NH, 256, 0, stream>>>(states, prev, logA);
    ssd_yoff_kernel<<<dim3(NCH, Bsz, NH), 256, 0, stream>>>(xbc, prev, logA, ybuf);

    gate_norm_kernel<<<BL, 256, 0, stream>>>(ybuf, proj, normw, ybf);

    // GEMM2: out = y @ W_out^T
    gemm_bf16_kernel<<<dim3(DM / 64, BL / 128), 256, 0, stream>>>(
        ybf, wout_bf, out, DIN, DM, DM);
}
